// Double_Deform_Block_10170482557069
// MI455X (gfx1250) — matrix operand order repaired and checked by reading, not compiled, not run on hardware
//
#include <hip/hip_runtime.h>
#include <hip/hip_bf16.h>

typedef __attribute__((ext_vector_type(16))) _Float16 v16h;
typedef __attribute__((ext_vector_type(8)))  _Float16 v8h;
typedef __attribute__((ext_vector_type(8)))  float    v8f;

#define BB 4
#define HH 128
#define WW 128
#define HWP (HH * WW)
#define KK 9

// ---------------------------------------------------------------------------
// fp32 -> f16 weight conversion, layout [O][C*9] (same flat order as OIHW)
// ---------------------------------------------------------------------------
__global__ __launch_bounds__(256) void convert_w_kernel(const float* __restrict__ in,
                                                        _Float16* __restrict__ out, int n) {
    int i = blockIdx.x * 256 + threadIdx.x;
    if (i < n) out[i] = (_Float16)in[i];
}

// ---------------------------------------------------------------------------
// Direct 3x3 conv, zero pad, for the 18-channel offset predictor.
// One thread per output element (b, oc, h, w).
// ---------------------------------------------------------------------------
template <int CIN>
__global__ __launch_bounds__(256) void conv_offset_kernel(const float* __restrict__ x,
                                                          const float* __restrict__ wgt,
                                                          const float* __restrict__ bias,
                                                          float* __restrict__ out) {
    const int i  = blockIdx.x * 256 + threadIdx.x;
    const int w  = i & (WW - 1);
    const int h  = (i >> 7) & (HH - 1);
    const int oc = (i / HWP) % (2 * KK);
    const int b  = i / (HWP * 2 * KK);

    float s = bias[oc];
    const float* xb = x + (size_t)b * CIN * HWP;
    const float* wo = wgt + (size_t)oc * CIN * 9;
    for (int c = 0; c < CIN; ++c) {
        const float* xc = xb + (size_t)c * HWP;
        const float* wc = wo + c * 9;
#pragma unroll
        for (int dy = 0; dy < 3; ++dy) {
            const int hy = h + dy - 1;
            if ((unsigned)hy < (unsigned)HH) {
#pragma unroll
                for (int dx = 0; dx < 3; ++dx) {
                    const int wx = w + dx - 1;
                    if ((unsigned)wx < (unsigned)WW) s += wc[dy * 3 + dx] * xc[hy * WW + wx];
                }
            }
        }
    }
    out[i] = s;
}

// ---------------------------------------------------------------------------
// Deformable conv: bilinear-sample im2col tile into LDS (f16), then
// per-wave 16x16x32 F16 WMMA GEMM: out[o, pix] = sum_kk W[o,kk] * S[kk,pix].
// Block = 8 wave32s, owns 16 consecutive pixels (same image row).
// Wave m computes output channels [16m, 16m+16).
// ---------------------------------------------------------------------------
template <int CIN, int CO>
__global__ __launch_bounds__(256) void deform_wmma_kernel(const float* __restrict__ x,
                                                          const float* __restrict__ off,
                                                          const _Float16* __restrict__ wh,
                                                          const float* __restrict__ bias,
                                                          float* __restrict__ y) {
    constexpr int KTOT = CIN * KK;     // 576 or 1152
    constexpr int STR  = KTOT + 8;     // padded row stride (halves), keeps 16B align
    __shared__ __align__(16) _Float16 smem[16 * STR];

    const int tid  = threadIdx.x;
    const int wave = tid >> 5;
    const int lane = tid & 31;

    const int tile = blockIdx.x;       // 16 pixels per tile
    const int p0   = tile * 16;
    const int b    = p0 / HWP;
    const int hw0  = p0 - b * HWP;
    const int h    = hw0 / WW;
    const int w0   = hw0 - h * WW;     // pixels w0..w0+15, one row

    // ---------------- sampling phase: build B matrix in LDS ----------------
    for (int pair = wave; pair < 16 * KK; pair += 8) {
        const int n  = pair / KK;      // pixel within tile
        const int k  = pair - n * KK;  // kernel tap 0..8
        const int wq = w0 + n;

        const float offy = off[((size_t)(b * 2 * KK + 2 * k)     * HH + h) * WW + wq];
        const float offx = off[((size_t)(b * 2 * KK + 2 * k + 1) * HH + h) * WW + wq];
        const float py = (float)h  - 1.0f + (float)(k / 3) + offy;
        const float px = (float)wq - 1.0f + (float)(k % 3) + offx;
        const float fy0 = floorf(py), fx0 = floorf(px);
        const int   y0  = (int)fy0,  x0  = (int)fx0;
        const float wy1 = py - fy0,  wx1 = px - fx0;
        const float wy0 = 1.0f - wy1, wx0 = 1.0f - wx1;

        const bool iy0 = (y0 >= 0) & (y0 < HH);
        const bool iy1 = (y0 + 1 >= 0) & (y0 + 1 < HH);
        const bool ix0 = (x0 >= 0) & (x0 < WW);
        const bool ix1 = (x0 + 1 >= 0) & (x0 + 1 < WW);
        const float v00 = (iy0 & ix0) ? wy0 * wx0 : 0.0f;
        const float v01 = (iy0 & ix1) ? wy0 * wx1 : 0.0f;
        const float v10 = (iy1 & ix0) ? wy1 * wx0 : 0.0f;
        const float v11 = (iy1 & ix1) ? wy1 * wx1 : 0.0f;

        const int y0c = min(max(y0, 0), HH - 1), y1c = min(max(y0 + 1, 0), HH - 1);
        const int x0c = min(max(x0, 0), WW - 1), x1c = min(max(x0 + 1, 0), WW - 1);
        const int i00 = y0c * WW + x0c, i01 = y0c * WW + x1c;
        const int i10 = y1c * WW + x0c, i11 = y1c * WW + x1c;

        const float* xb = x + (size_t)b * CIN * HWP;
        for (int c = lane; c < CIN; c += 32) {
            const float* xc = xb + (size_t)c * HWP;
            const float s = xc[i00] * v00 + xc[i01] * v01 + xc[i10] * v10 + xc[i11] * v11;
            smem[n * STR + c * KK + k] = (_Float16)s;   // B[kk = c*9+k][n]
        }
    }
    __syncthreads();

    // ---------------- WMMA GEMM phase ----------------
    // A frag (16x32 f16): lane holds row M = lane%16; halves i<8 -> K=base+i,
    // i>=8 -> K=16+base+(i-8), base = lane<16 ? 0 : 8.
    // B frag (32x16 f16): lane holds col N = lane%16; halves i<8 -> K=kgrp+i,
    // i>=8 -> K=16+kgrp+(i-8), kgrp = lane<16 ? 0 : 8.
    const int mrow  = (wave << 4) + (lane & 15);
    const int abase = (lane < 16) ? 0 : 8;
    const int kgrp  = (lane < 16) ? 0 : 8;
    const _Float16* arow = wh + (size_t)mrow * KTOT + abase;
    const _Float16* brow = smem + (size_t)(lane & 15) * STR + kgrp;

    v8f acc = {};
    for (int kb = 0; kb < KTOT; kb += 32) {
        union { v16h v; v8h h[2]; } A, Bm;
        A.h[0]  = *(const v8h*)(arow + kb);
        A.h[1]  = *(const v8h*)(arow + kb + 16);
        Bm.h[0] = *(const v8h*)(brow + kb);
        Bm.h[1] = *(const v8h*)(brow + kb + 16);
        acc = __builtin_amdgcn_wmma_f32_16x16x32_f16(false, A.v, false, Bm.v,
                                                     (short)0, acc, false, false);
    }

    // D layout: VGPR v -> (M = v + (lane<16?0:8), N = lane%16)
    const int nn    = lane & 15;
    const int hw    = hw0 + nn;
    const int obase = (wave << 4) + ((lane < 16) ? 0 : 8);
    float* yb = y + (size_t)b * CO * HWP + hw;
#pragma unroll
    for (int v = 0; v < 8; ++v) {
        const int o = obase + v;
        yb[(size_t)o * HWP] = acc[v] + bias[o];
    }
}

// ---------------------------------------------------------------------------
// Per-channel mean/var over (B,H,W) — deterministic block reduction.
// stats[0..CO) = mean, stats[CO..2CO) = var
// ---------------------------------------------------------------------------
template <int CO>
__global__ __launch_bounds__(256) void bn_stats_kernel(const float* __restrict__ y,
                                                       float* __restrict__ stats) {
    const int ch = blockIdx.x;
    __shared__ float ssum[256], ssq[256];
    float s = 0.0f, q = 0.0f;
    for (int idx = threadIdx.x; idx < BB * HWP; idx += 256) {
        const int b = idx / HWP, hw = idx - b * HWP;
        const float v = y[((size_t)b * CO + ch) * HWP + hw];
        s += v; q += v * v;
    }
    ssum[threadIdx.x] = s; ssq[threadIdx.x] = q;
    __syncthreads();
    for (int st = 128; st > 0; st >>= 1) {
        if (threadIdx.x < st) {
            ssum[threadIdx.x] += ssum[threadIdx.x + st];
            ssq[threadIdx.x]  += ssq[threadIdx.x + st];
        }
        __syncthreads();
    }
    if (threadIdx.x == 0) {
        const float inv  = 1.0f / (float)(BB * HWP);
        const float mean = ssum[0] * inv;
        stats[ch]      = mean;
        stats[CO + ch] = ssq[0] * inv - mean * mean;
    }
}

template <int CO>
__global__ __launch_bounds__(256) void bn_apply_kernel(const float* __restrict__ y,
                                                       const float* __restrict__ stats,
                                                       const float* __restrict__ g,
                                                       const float* __restrict__ be,
                                                       float* __restrict__ out) {
    const size_t i  = (size_t)blockIdx.x * 256 + threadIdx.x;
    const int    ch = (int)((i / HWP) % CO);
    const float mean = stats[ch], var = stats[CO + ch];
    const float v = (y[i] - mean) * rsqrtf(var + 1e-5f) * g[ch] + be[ch];
    out[i] = v > 0.0f ? v : 0.0f;
}

// ---------------------------------------------------------------------------
extern "C" void kernel_launch(void* const* d_in, const int* in_sizes, int n_in,
                              void* d_out, int out_size, void* d_ws, size_t ws_size,
                              hipStream_t stream) {
    const float* x      = (const float*)d_in[0];
    const float* w_off1 = (const float*)d_in[1];
    const float* b_off1 = (const float*)d_in[2];
    const float* w1     = (const float*)d_in[3];
    const float* b1     = (const float*)d_in[4];
    const float* g1     = (const float*)d_in[5];
    const float* be1    = (const float*)d_in[6];
    const float* w_off2 = (const float*)d_in[7];
    const float* b_off2 = (const float*)d_in[8];
    const float* w2     = (const float*)d_in[9];
    const float* b2     = (const float*)d_in[10];
    const float* g2     = (const float*)d_in[11];
    const float* be2    = (const float*)d_in[12];
    float* out = (float*)d_out;

    // workspace carve-up (256B aligned)
    char* ws = (char*)d_ws;
    size_t cur = 0;
    auto carve = [&](size_t bytes) {
        char* p = ws + cur;
        cur += (bytes + 255) & ~(size_t)255;
        return p;
    };
    float*     off_buf = (float*)    carve(sizeof(float) * BB * 2 * KK * HWP);   // 4.72 MB
    float*     y_buf   = (float*)    carve(sizeof(float) * BB * 128 * HWP);      // 33.5 MB
    float*     yn_buf  = (float*)    carve(sizeof(float) * BB * 128 * HWP);      // 33.5 MB
    _Float16*  wh1     = (_Float16*) carve(sizeof(_Float16) * 128 * 64 * 9);
    _Float16*  wh2     = (_Float16*) carve(sizeof(_Float16) * 128 * 128 * 9);
    float*     stats   = (float*)    carve(sizeof(float) * 256);

    const int nw1 = 128 * 64 * 9, nw2 = 128 * 128 * 9;
    convert_w_kernel<<<(nw1 + 255) / 256, 256, 0, stream>>>(w1, wh1, nw1);
    convert_w_kernel<<<(nw2 + 255) / 256, 256, 0, stream>>>(w2, wh2, nw2);

    const int off_grid = BB * 2 * KK * HWP / 256;   // 4608
    const int dfm_grid = BB * HWP / 16;             // 4096 pixel tiles
    const int ew_grid  = BB * 128 * HWP / 256;      // 32768

    // ---- round 1: x(64ch) -> y(128ch) ----
    conv_offset_kernel<64><<<off_grid, 256, 0, stream>>>(x, w_off1, b_off1, off_buf);
    deform_wmma_kernel<64, 128><<<dfm_grid, 256, 0, stream>>>(x, off_buf, wh1, b1, y_buf);
    bn_stats_kernel<128><<<128, 256, 0, stream>>>(y_buf, stats);
    bn_apply_kernel<128><<<ew_grid, 256, 0, stream>>>(y_buf, stats, g1, be1, yn_buf);

    // ---- round 2: yn(128ch) -> out(128ch) ----
    conv_offset_kernel<128><<<off_grid, 256, 0, stream>>>(yn_buf, w_off2, b_off2, off_buf);
    deform_wmma_kernel<128, 128><<<dfm_grid, 256, 0, stream>>>(yn_buf, off_buf, wh2, b2, y_buf);
    bn_stats_kernel<128><<<128, 256, 0, stream>>>(y_buf, stats);
    bn_apply_kernel<128><<<ew_grid, 256, 0, stream>>>(y_buf, stats, g2, be2, out);
}
